// GCN_large_57105885167694
// MI455X (gfx1250) — matrix operand order repaired and checked by reading, not compiled, not run on hardware
//
#include <hip/hip_runtime.h>
#include <hip/hip_bf16.h>

typedef __attribute__((ext_vector_type(16))) __bf16 v16bf;
typedef __attribute__((ext_vector_type(8)))  float  v8f;

#define N_NODES 50000
#define BN_EPS 1e-5f

// ---------------------------------------------------------------------------
// Degree / dinv precompute
// ---------------------------------------------------------------------------
__global__ void gcn_deg_init(float* __restrict__ deg, int n) {
    int i = blockIdx.x * blockDim.x + threadIdx.x;
    if (i < n) deg[i] = 1.0f;   // self-loop weight
}

__global__ void gcn_deg_accum(const int* __restrict__ dst,
                              const float* __restrict__ ew,
                              float* __restrict__ deg, int E) {
    int e = blockIdx.x * blockDim.x + threadIdx.x;
    if (e < E) atomicAdd(&deg[dst[e]], ew[e]);
}

__global__ void gcn_dinv_fin(float* __restrict__ deg, int n) {
    int i = blockIdx.x * blockDim.x + threadIdx.x;
    if (i < n) {
        float d = deg[i];
        deg[i] = (d > 0.0f) ? rsqrtf(d) : 0.0f;
    }
}

// ---------------------------------------------------------------------------
// Weight prep: Wt[Fpad x 128] = transpose(W[128 x Fout]), zero-padded columns.
// ---------------------------------------------------------------------------
__global__ void gcn_weight_prep(const float* __restrict__ W,
                                float* __restrict__ Wt, int Fout, int Fpad) {
    int t = blockIdx.x * blockDim.x + threadIdx.x;   // t = n*128 + k
    if (t >= Fpad * 128) return;
    int n = t >> 7;
    int k = t & 127;
    Wt[t] = (n < Fout) ? W[k * Fout + n] : 0.0f;
}

// ---------------------------------------------------------------------------
// GEMM: Y[M x Fpad] = X[M x 128] @ Wt^T, bf16 WMMA, f32 accumulate.
// Each wave computes a 16 x (TNG*16) strip: the A fragment is loaded once per
// k-block and reused across TNG B fragments / accumulators.
// K = 128 -> 4 k-blocks x TNG WMMAs. Guard-free: Fpad multiple of 16*TNG/groups,
// M multiple of 16.
// ---------------------------------------------------------------------------
template <int TNG>
__global__ __launch_bounds__(256) void gcn_gemm_wmma(
        const float* __restrict__ X, const float* __restrict__ Wt,
        float* __restrict__ Y, int M, int Fpad) {
    const int wave = threadIdx.x >> 5;
    const int lane = threadIdx.x & 31;
    const int groupsN = (Fpad >> 4) / TNG;
    const int totalWork = (M >> 4) * groupsN;
    const int work = blockIdx.x * 8 + wave;   // wave-uniform
    if (work >= totalWork) return;            // uniform branch: EXEC stays all-1s

    const int tm = work / groupsN;
    const int g  = work % groupsN;
    const int half = lane >> 4;
    const int l15  = lane & 15;

    const float* __restrict__ xrow  = X  + (unsigned)((tm << 4) + l15) * 128u;
    const float* __restrict__ wbase = Wt + (unsigned)(g * TNG * 16 + l15) * 128u;

    v8f acc[TNG];
    #pragma unroll
    for (int t = 0; t < TNG; ++t) acc[t] = (v8f){};

    #pragma unroll
    for (int kb = 0; kb < 128; kb += 32) {
        // A fragment 16x32 bf16: lanes 0-15 K=kb+[0..7],[16..23];
        //                        lanes 16-31 K=kb+[8..15],[24..31]
        const float4* xr = reinterpret_cast<const float4*>(xrow + kb + half * 8);
        float4 x0 = xr[0];   // K offsets 0..3
        float4 x1 = xr[1];   // 4..7
        float4 x2 = xr[4];   // 16..19
        float4 x3 = xr[5];   // 20..23
        v16bf a;
        a[0]=(__bf16)x0.x; a[1]=(__bf16)x0.y; a[2]=(__bf16)x0.z; a[3]=(__bf16)x0.w;
        a[4]=(__bf16)x1.x; a[5]=(__bf16)x1.y; a[6]=(__bf16)x1.z; a[7]=(__bf16)x1.w;
        a[8]=(__bf16)x2.x; a[9]=(__bf16)x2.y; a[10]=(__bf16)x2.z; a[11]=(__bf16)x2.w;
        a[12]=(__bf16)x3.x; a[13]=(__bf16)x3.y; a[14]=(__bf16)x3.z; a[15]=(__bf16)x3.w;

        #pragma unroll
        for (int t = 0; t < TNG; ++t) {
            // B fragment 32x16 bf16: lanes 0-15 K=kb+[0..7],[16..23]; lanes 16-31 K=kb+[8..15],[24..31]
            const float4* wr = reinterpret_cast<const float4*>(
                wbase + (unsigned)(t * 16 * 128) + kb + half * 8);
            float4 w0 = wr[0], w1 = wr[1], w2 = wr[4], w3 = wr[5];
            v16bf b;
            b[0]=(__bf16)w0.x; b[1]=(__bf16)w0.y; b[2]=(__bf16)w0.z; b[3]=(__bf16)w0.w;
            b[4]=(__bf16)w1.x; b[5]=(__bf16)w1.y; b[6]=(__bf16)w1.z; b[7]=(__bf16)w1.w;
            b[8]=(__bf16)w2.x; b[9]=(__bf16)w2.y; b[10]=(__bf16)w2.z; b[11]=(__bf16)w2.w;
            b[12]=(__bf16)w3.x; b[13]=(__bf16)w3.y; b[14]=(__bf16)w3.z; b[15]=(__bf16)w3.w;

            acc[t] = __builtin_amdgcn_wmma_f32_16x16x32_bf16(
                         false, a, false, b, (short)0, acc[t], false, false);
        }
    }

    // C layout: VGPR r -> M = r + 8*half, N = lane&15
    const unsigned row0 = (unsigned)((tm << 4) + half * 8);
    #pragma unroll
    for (int t = 0; t < TNG; ++t) {
        float* __restrict__ yout =
            Y + row0 * (unsigned)Fpad + (unsigned)((g * TNG + t) * 16) + (unsigned)l15;
        #pragma unroll
        for (int r = 0; r < 8; ++r)
            yout[(unsigned)(r * Fpad)] = acc[t][r];
    }
}

// ---------------------------------------------------------------------------
// H[i,f] = bias[f] + dinv[i]^2 * XW[i,f]   (self-loop term + bias, full init)
// ---------------------------------------------------------------------------
template <int F, int LD>
__global__ void gcn_self_init(const float* __restrict__ XW,
                              const float* __restrict__ dinv,
                              const float* __restrict__ bias,
                              float* __restrict__ H, int n) {
    unsigned t = blockIdx.x * blockDim.x + threadIdx.x;
    unsigned total = (unsigned)n * F;
    if (t >= total) return;
    unsigned i = t / F, f = t % F;
    float di = dinv[i];
    H[t] = bias[f] + di * di * XW[i * LD + f];
}

// ---------------------------------------------------------------------------
// Edge scatter: H[dst] += dinv[src]*w*dinv[dst] * XW[src], float4 per thread
// ---------------------------------------------------------------------------
template <int F, int LD>
__global__ void gcn_scatter(const int* __restrict__ src, const int* __restrict__ dst,
                            const float* __restrict__ ew, const float* __restrict__ dinv,
                            const float* __restrict__ XW, float* __restrict__ H,
                            int E) {
    constexpr int chunks = F >> 2;
    unsigned t = blockIdx.x * blockDim.x + threadIdx.x;
    unsigned total = (unsigned)E * chunks;
    if (t >= total) return;
    unsigned e = t / chunks;
    unsigned c = (t % chunks) << 2;
    int s = src[e], d = dst[e];
    float nrm = dinv[s] * ew[e] * dinv[d];
    const float4 v = *reinterpret_cast<const float4*>(XW + (unsigned)s * LD + c);
    float* out = H + (unsigned)d * F + c;
    atomicAdd(out + 0, nrm * v.x);
    atomicAdd(out + 1, nrm * v.y);
    atomicAdd(out + 2, nrm * v.z);
    atomicAdd(out + 3, nrm * v.w);
}

// ---------------------------------------------------------------------------
// BatchNorm (F = 128): stats accumulate (LDS-reduced), finalize, apply+ReLU
// ---------------------------------------------------------------------------
__global__ void gcn_bn_zero(float* __restrict__ stats) {
    if (threadIdx.x < 256) stats[threadIdx.x] = 0.0f;   // gsum[128] + gsq[128]
}

__global__ __launch_bounds__(256) void gcn_bn_accum(const float* __restrict__ H, int n,
                                                    float* __restrict__ gsum,
                                                    float* __restrict__ gsq) {
    __shared__ float ls[256], lq[256];
    const int f   = threadIdx.x & 127;
    const int sub = threadIdx.x >> 7;    // 0 or 1
    float s = 0.0f, q = 0.0f;
    for (int r = blockIdx.x * 2 + sub; r < n; r += gridDim.x * 2) {
        float v = H[(unsigned)r * 128 + f];
        s += v; q += v * v;
    }
    ls[threadIdx.x] = s; lq[threadIdx.x] = q;
    __syncthreads();
    if (sub == 0) {
        s += ls[threadIdx.x + 128];
        q += lq[threadIdx.x + 128];
        atomicAdd(&gsum[f], s);
        atomicAdd(&gsq[f], q);
    }
}

__global__ void gcn_bn_finalize(const float* __restrict__ gsum, const float* __restrict__ gsq,
                                const float* __restrict__ gamma, const float* __restrict__ beta,
                                float* __restrict__ scale, float* __restrict__ shift, int n) {
    int f = threadIdx.x;
    if (f < 128) {
        float invN = 1.0f / (float)n;
        float mu  = gsum[f] * invN;
        float var = gsq[f] * invN - mu * mu;
        float sc  = gamma[f] * rsqrtf(var + BN_EPS);
        scale[f] = sc;
        shift[f] = beta[f] - mu * sc;
    }
}

__global__ void gcn_bn_apply_relu(float* __restrict__ H,
                                  const float* __restrict__ scale,
                                  const float* __restrict__ shift, int n) {
    unsigned t = blockIdx.x * blockDim.x + threadIdx.x;
    unsigned total = (unsigned)n * 128;
    if (t >= total) return;
    unsigned f = t & 127;
    float v = fmaf(H[t], scale[f], shift[f]);
    H[t] = fmaxf(v, 0.0f);
}

// ---------------------------------------------------------------------------
// log_softmax over 40 classes, one wave32 per row, in place
// ---------------------------------------------------------------------------
__global__ __launch_bounds__(128) void gcn_log_softmax40(float* __restrict__ Y, int n) {
    const int wave = threadIdx.x >> 5;
    const int lane = threadIdx.x & 31;
    int row = blockIdx.x * 4 + wave;
    if (row >= n) return;
    float* y = Y + (unsigned)row * 40;
    float a = y[lane];                                    // elements 0..31
    float b = (lane < 8) ? y[lane + 32] : -__builtin_inff();
    float m = fmaxf(a, b);
    #pragma unroll
    for (int off = 16; off; off >>= 1) m = fmaxf(m, __shfl_xor(m, off, 32));
    float se = __expf(a - m) + ((lane < 8) ? __expf(b - m) : 0.0f);
    #pragma unroll
    for (int off = 16; off; off >>= 1) se += __shfl_xor(se, off, 32);
    float lse = m + __logf(se);
    y[lane] = a - lse;
    if (lane < 8) y[lane + 32] = b - lse;
}

// ---------------------------------------------------------------------------
// Host-side orchestration
// ---------------------------------------------------------------------------
template <int F, int LD, int TNG>
static inline void run_gcn_layer(const float* Xin, const float* W, const float* bias,
                                 float* Wt, float* XW, float* H, const float* dinv,
                                 const int* src, const int* dst, const float* ew, int E,
                                 hipStream_t stream) {
    const int M = N_NODES;
    gcn_weight_prep<<<(LD * 128 + 255) / 256, 256, 0, stream>>>(W, Wt, F, LD);

    int totalWork = (M >> 4) * ((LD >> 4) / TNG);
    gcn_gemm_wmma<TNG><<<(totalWork + 7) / 8, 256, 0, stream>>>(Xin, Wt, XW, M, LD);

    unsigned initTotal = (unsigned)M * F;
    gcn_self_init<F, LD><<<(initTotal + 255) / 256, 256, 0, stream>>>(XW, dinv, bias, H, M);

    unsigned scatTotal = (unsigned)E * (F >> 2);
    gcn_scatter<F, LD><<<(scatTotal + 255) / 256, 256, 0, stream>>>(src, dst, ew, dinv, XW, H, E);
}

extern "C" void kernel_launch(void* const* d_in, const int* in_sizes, int n_in,
                              void* d_out, int out_size, void* d_ws, size_t ws_size,
                              hipStream_t stream) {
    const float* x      = (const float*)d_in[0];
    const int*   ei     = (const int*)d_in[1];
    const float* ew     = (const float*)d_in[2];
    const float* W1     = (const float*)d_in[3];
    const float* b1     = (const float*)d_in[4];
    const float* gamma1 = (const float*)d_in[5];
    const float* beta1  = (const float*)d_in[6];
    const float* W2     = (const float*)d_in[7];
    const float* b2     = (const float*)d_in[8];
    const float* gamma2 = (const float*)d_in[9];
    const float* beta2  = (const float*)d_in[10];
    const float* W3     = (const float*)d_in[11];
    const float* b3     = (const float*)d_in[12];

    const int E = in_sizes[1] / 2;
    const int* src = ei;
    const int* dst = ei + E;
    const int N = N_NODES;

    // Workspace layout
    char* ws = (char*)d_ws;
    float* dinv  = (float*)ws;                                              // 50176 floats
    float* bufA  = (float*)(ws + (size_t)50176 * 4);                        // N*128 (GEMM out)
    float* bufB  = (float*)(ws + (size_t)50176 * 4 + (size_t)N * 128 * 4);  // N*128 (H)
    float* stats = (float*)(ws + (size_t)50176 * 4 + 2 * (size_t)N * 128 * 4);
    float* gsum  = stats;          // 128
    float* gsq   = stats + 128;    // 128
    float* scale = stats + 256;    // 128
    float* shift = stats + 384;    // 128
    float* Wt    = stats + 512;    // up to 128*128 floats (transposed, padded weights)

    // deg -> dinv
    gcn_deg_init<<<(N + 255) / 256, 256, 0, stream>>>(dinv, N);
    gcn_deg_accum<<<(E + 255) / 256, 256, 0, stream>>>(dst, ew, dinv, E);
    gcn_dinv_fin<<<(N + 255) / 256, 256, 0, stream>>>(dinv, N);

    // Layer 1: x[N,128] -> bufB[N,128], BN+ReLU in place
    run_gcn_layer<128, 128, 4>(x, W1, b1, Wt, bufA, bufB, dinv, src, dst, ew, E, stream);
    gcn_bn_zero<<<1, 256, 0, stream>>>(stats);
    gcn_bn_accum<<<128, 256, 0, stream>>>(bufB, N, gsum, gsq);
    gcn_bn_finalize<<<1, 128, 0, stream>>>(gsum, gsq, gamma1, beta1, scale, shift, N);
    gcn_bn_apply_relu<<<((unsigned)N * 128 + 255) / 256, 256, 0, stream>>>(bufB, scale, shift, N);

    // Layer 2: bufB -> bufB (GEMM consumes bufB into bufA before H init rewrites it)
    run_gcn_layer<128, 128, 4>(bufB, W2, b2, Wt, bufA, bufB, dinv, src, dst, ew, E, stream);
    gcn_bn_zero<<<1, 256, 0, stream>>>(stats);
    gcn_bn_accum<<<128, 256, 0, stream>>>(bufB, N, gsum, gsq);
    gcn_bn_finalize<<<1, 128, 0, stream>>>(gsum, gsq, gamma2, beta2, scale, shift, N);
    gcn_bn_apply_relu<<<((unsigned)N * 128 + 255) / 256, 256, 0, stream>>>(bufB, scale, shift, N);

    // Layer 3: bufB -> d_out[N,40]; GEMM into padded bufA (ld=48, 3 n-tiles/wave)
    run_gcn_layer<40, 48, 3>(bufB, W3, b3, Wt, bufA, (float*)d_out, dinv, src, dst, ew, E, stream);

    // log_softmax in place on d_out
    gcn_log_softmax40<<<(N + 3) / 4, 128, 0, stream>>>((float*)d_out, N);
}
